// WeightedGATLayer_90005334655074
// MI455X (gfx1250) — matrix operand order repaired and checked by reading, not compiled, not run on hardware
//
#include <hip/hip_runtime.h>
#include <hip/hip_bf16.h>

// ---------- types for WMMA fragments ----------
typedef __bf16 bf16_t;
typedef bf16_t v16bf __attribute__((ext_vector_type(16)));
typedef float  v8f   __attribute__((ext_vector_type(8)));
typedef unsigned int v4u __attribute__((ext_vector_type(4)));

#define N_NODES 50000
#define N_EDGES 1600000
#define IN_F    256
#define OUT_C   128   // OUT_FEATS * NUM_HEADS
#define N_HEADS 4
#define HDIM    32
#define NEG_SLOPE 0.2f

// ---------- helpers ----------
__device__ __forceinline__ unsigned short f2bf_rne(float f) {
    unsigned u = __float_as_uint(f);
    u += 0x7FFFu + ((u >> 16) & 1u);       // round-to-nearest-even
    return (unsigned short)(u >> 16);
}

union Frag32B { v4u u[2]; v16bf v; };
__device__ __forceinline__ v16bf load2x16(const unsigned short* p0,
                                          const unsigned short* p1) {
    Frag32B t;
    t.u[0] = *(const v4u*)p0;   // elements 0..7
    t.u[1] = *(const v4u*)p1;   // elements 8..15
    return t.v;
}

// order-preserving float<->uint mapping for atomicMax (mapped(finite) > 0)
__device__ __forceinline__ unsigned fmap(float f) {
    unsigned b = __float_as_uint(f);
    return (b & 0x80000000u) ? ~b : (b | 0x80000000u);
}
__device__ __forceinline__ float funmap(unsigned m) {
    return (m & 0x80000000u) ? __uint_as_float(m ^ 0x80000000u)
                             : __uint_as_float(~m);
}

// ---------- kernel 1: feat f32 -> bf16 ----------
__global__ __launch_bounds__(256) void k_conv_feat(const float* __restrict__ feat,
                                                   unsigned short* __restrict__ featB) {
    int t = blockIdx.x * 256 + threadIdx.x;
    if (t < N_NODES * IN_F) featB[t] = f2bf_rne(feat[t]);
}

// ---------- kernel 2: W [256,128] f32 -> Wt [128,256] bf16 ----------
__global__ __launch_bounds__(256) void k_conv_w(const float* __restrict__ W,
                                                unsigned short* __restrict__ WtB) {
    int t = blockIdx.x * 256 + threadIdx.x;   // t = n*256 + k
    if (t < OUT_C * IN_F) {
        int n = t >> 8, k = t & 255;
        WtB[t] = f2bf_rne(W[k * OUT_C + n]);
    }
}

// ---------- kernel 3: ft = featB @ WtB^T via v_wmma_f32_16x16x32_bf16 ----------
// block = 256 threads = 8 waves; wave w computes 16x16 tile at cols [16w,16w+16)
__global__ __launch_bounds__(256) void k_gemm_wmma(const unsigned short* __restrict__ featB,
                                                   const unsigned short* __restrict__ WtB,
                                                   float* __restrict__ ft) {
    const int lane  = threadIdx.x & 31;
    const int ntile = threadIdx.x >> 5;          // 0..7
    const int mtile = blockIdx.x;                // 0..3124 (50000/16)
    const int l15   = lane & 15;
    const int hi    = lane >> 4;                 // 0 or 1

    const unsigned short* aRow = featB + (size_t)(mtile * 16 + l15) * IN_F;
    const unsigned short* bRow = WtB   + (size_t)(ntile * 16 + l15) * IN_F;
    const int aSel = hi << 3;   // A: lanes16-31 start at K+8 (ISA 16-bit A layout)
    const int bSel = hi << 3;   // B: lanes16-31 start at K+8 (same layout as A)

    v8f acc = {};
#pragma unroll
    for (int k0 = 0; k0 < IN_F; k0 += 32) {
        // A elems 0..7 = K kb..kb+7, elems 8..15 = K kb+16..kb+23
        v16bf a = load2x16(aRow + k0 + aSel, aRow + k0 + aSel + 16);
        // B elems 0..7 = K kb..kb+7, elems 8..15 = K kb+16..kb+23
        v16bf b = load2x16(bRow + k0 + bSel, bRow + k0 + bSel + 16);
        acc = __builtin_amdgcn_wmma_f32_16x16x32_bf16(
            /*neg_a=*/false, a, /*neg_b=*/false, b,
            /*c_mod=*/(short)0, acc, /*reuse_a=*/false, /*reuse_b=*/false);
    }

    // C layout: acc[r] -> (M = r + 8*hi, N = l15)
    float* outp = ft + (size_t)(mtile * 16 + (hi << 3)) * OUT_C + ntile * 16 + l15;
#pragma unroll
    for (int r = 0; r < 8; ++r) outp[(size_t)r * OUT_C] = acc[r];
}

// ---------- kernel 4: per-node attention logits el/er ----------
// block = 128 = 4 waves; wave h reduces over D=32 for node blockIdx.x
__global__ __launch_bounds__(128) void k_logits(const float* __restrict__ ft,
                                                const float* __restrict__ attn_l,
                                                const float* __restrict__ attn_r,
                                                float* __restrict__ el,
                                                float* __restrict__ er) {
    int n = blockIdx.x;
    int h = threadIdx.x >> 5;
    int d = threadIdx.x & 31;
    float v = ft[(size_t)n * OUT_C + h * HDIM + d];
    float l = v * attn_l[h * HDIM + d];
    float r = v * attn_r[h * HDIM + d];
#pragma unroll
    for (int m = 16; m; m >>= 1) {
        l += __shfl_xor(l, m, 32);
        r += __shfl_xor(r, m, 32);
    }
    if (d == 0) { el[n * N_HEADS + h] = l; er[n * N_HEADS + h] = r; }
}

__device__ __forceinline__ float edge_score(const float* el, const float* er,
                                            const float* ew, float lam,
                                            int s, int d, int h, int e) {
    float v = el[s * N_HEADS + h] + er[d * N_HEADS + h] + lam * ew[e];
    return v > 0.f ? v : NEG_SLOPE * v;
}

// ---------- kernel 5: segment max over incoming edges ----------
__global__ __launch_bounds__(256) void k_edge_max(const float* __restrict__ el,
                                                  const float* __restrict__ er,
                                                  const float* __restrict__ ew,
                                                  const int* __restrict__ src,
                                                  const int* __restrict__ dst,
                                                  const float* __restrict__ lam,
                                                  unsigned* __restrict__ maxU) {
    int t = blockIdx.x * 256 + threadIdx.x;
    if (t >= N_EDGES * N_HEADS) return;
    int e = t >> 2, h = t & 3;
    int s = src[e], d = dst[e];
    float v = edge_score(el, er, ew, lam[0], s, d, h, e);
    atomicMax(&maxU[d * N_HEADS + h], fmap(v));
}

// ---------- kernel 6: softmax denominator ----------
__global__ __launch_bounds__(256) void k_edge_denom(const float* __restrict__ el,
                                                    const float* __restrict__ er,
                                                    const float* __restrict__ ew,
                                                    const int* __restrict__ src,
                                                    const int* __restrict__ dst,
                                                    const float* __restrict__ lam,
                                                    const unsigned* __restrict__ maxU,
                                                    float* __restrict__ denom) {
    int t = blockIdx.x * 256 + threadIdx.x;
    if (t >= N_EDGES * N_HEADS) return;
    int e = t >> 2, h = t & 3;
    int s = src[e], d = dst[e];
    float v = edge_score(el, er, ew, lam[0], s, d, h, e);
    float m = funmap(maxU[d * N_HEADS + h]);
    atomicAdd(&denom[d * N_HEADS + h], __expf(v - m));
}

// ---------- kernel 7: weighted message scatter ----------
// block = 128 per edge: wave h handles all 32 d's -> coalesced f32 atomics
__global__ __launch_bounds__(128) void k_scatter(const float* __restrict__ ft,
                                                 const float* __restrict__ el,
                                                 const float* __restrict__ er,
                                                 const float* __restrict__ ew,
                                                 const int* __restrict__ src,
                                                 const int* __restrict__ dst,
                                                 const float* __restrict__ lam,
                                                 const unsigned* __restrict__ maxU,
                                                 const float* __restrict__ denom,
                                                 float* __restrict__ out) {
    int e = blockIdx.x;
    int h = threadIdx.x >> 5;
    int d = threadIdx.x & 31;
    int s = src[e], dn = dst[e];            // uniform -> scalar loads
    float v = edge_score(el, er, ew, lam[0], s, dn, h, e);
    float m = funmap(maxU[dn * N_HEADS + h]);
    float a = __expf(v - m) / denom[dn * N_HEADS + h];
    float msg = ft[(size_t)s * OUT_C + h * HDIM + d] * a;
    atomicAdd(&out[(size_t)dn * OUT_C + h * HDIM + d], msg);
}

extern "C" void kernel_launch(void* const* d_in, const int* in_sizes, int n_in,
                              void* d_out, int out_size, void* d_ws, size_t ws_size,
                              hipStream_t stream) {
    const float* feat = (const float*)d_in[0];
    const float* ew   = (const float*)d_in[1];
    const int*   src  = (const int*)d_in[2];
    const int*   dst  = (const int*)d_in[3];
    const float* W    = (const float*)d_in[4];
    const float* al   = (const float*)d_in[5];
    const float* ar   = (const float*)d_in[6];
    const float* lam  = (const float*)d_in[7];
    float* out = (float*)d_out;

    // workspace carve-up (256B aligned)
    char* ws = (char*)d_ws;
    size_t o = 0;
    auto carve = [&](size_t bytes) { char* p = ws + o; o += (bytes + 255) & ~(size_t)255; return p; };
    unsigned short* featB = (unsigned short*)carve((size_t)N_NODES * IN_F * 2);   // 25.6 MB
    unsigned short* WtB   = (unsigned short*)carve((size_t)OUT_C * IN_F * 2);     // 64 KB
    float*    ft    = (float*)   carve((size_t)N_NODES * OUT_C * 4);              // 25.6 MB
    float*    el    = (float*)   carve((size_t)N_NODES * N_HEADS * 4);
    float*    er    = (float*)   carve((size_t)N_NODES * N_HEADS * 4);
    unsigned* maxU  = (unsigned*)carve((size_t)N_NODES * N_HEADS * 4);
    float*    denom = (float*)   carve((size_t)N_NODES * N_HEADS * 4);

    // zero init (memsets are graph-capture safe); mapped-max init 0 < mapped(any finite)
    hipMemsetAsync(maxU,  0, (size_t)N_NODES * N_HEADS * 4, stream);
    hipMemsetAsync(denom, 0, (size_t)N_NODES * N_HEADS * 4, stream);
    hipMemsetAsync(out,   0, (size_t)out_size * sizeof(float), stream);

    // 1-2: bf16 conversions
    k_conv_feat<<<(N_NODES * IN_F + 255) / 256, 256, 0, stream>>>(feat, featB);
    k_conv_w<<<(OUT_C * IN_F + 255) / 256, 256, 0, stream>>>(W, WtB);

    // 3: WMMA projection GEMM (3125 M-tiles x 8 waves covering all 128 cols)
    k_gemm_wmma<<<N_NODES / 16, 256, 0, stream>>>(featB, WtB, ft);

    // 4: attention logits
    k_logits<<<N_NODES, 128, 0, stream>>>(ft, al, ar, el, er);

    // 5-6: edge softmax statistics
    int eh_blocks = (N_EDGES * N_HEADS + 255) / 256;
    k_edge_max<<<eh_blocks, 256, 0, stream>>>(el, er, ew, src, dst, lam, maxU);
    k_edge_denom<<<eh_blocks, 256, 0, stream>>>(el, er, ew, src, dst, lam, maxU, denom);

    // 7: weighted message passing (coalesced atomics, one block per edge)
    k_scatter<<<N_EDGES, 128, 0, stream>>>(ft, el, er, ew, src, dst, lam, maxU, denom, out);
}
